// RecurEncoder_55430847922541
// MI455X (gfx1250) — compile-verified, matrix operand order checked by reading, not run on hardware
//
#include <hip/hip_runtime.h>

// 2-layer LSTM (PyTorch gate order i,f,g,o) for MI455X / gfx1250.
// Strategy: batch-parallel persistent workgroups (16 rows each), bf16 WMMA
// GEMMs with fused [h|x] @ [Whh|Wih]^T per step, fp32 cell state in regs,
// double-buffered LDS h, weights pre-packed into WMMA B-fragment layout and
// re-streamed from L2 every step. An opaque SGPR offset (not an opaque
// pointer!) defeats LICM while preserving global address-space provenance,
// so weight loads stay global_load_b128 (not flat_load, not scratch spills).

typedef unsigned short u16;
typedef __attribute__((ext_vector_type(16))) __bf16       v16bf;
typedef __attribute__((ext_vector_type(8)))  float        v8f;
typedef __attribute__((ext_vector_type(4)))  unsigned int v4u;

#define BATCH  256
#define TSTEPS 512
#define NFEAT  64
#define HID1   256
#define GATE1  1024
#define EMB2   128
#define GATE2  512
#define KTOT1  320   // 256 (h) + 64 (x)
#define KTOT2  384   // 128 (h) + 256 (h1 input)
#define KT1    10    // KTOT1 / 32
#define KT2    12    // KTOT2 / 32
#define LDST1  264   // HID1 + 8 halfword pad (bank-conflict-free row stride)
#define LDST2  136   // EMB2 + 8
#define MTILE  16

union FragU { v16bf bf; v4u u[2]; };

__device__ __forceinline__ u16 f2bf(float f) {
  unsigned int u = __float_as_uint(f);
  u += 0x7FFFu + ((u >> 16) & 1u);   // round-to-nearest-even
  return (u16)(u >> 16);
}
__device__ __forceinline__ float sigf(float x) { return 1.0f / (1.0f + __expf(-x)); }
__device__ __forceinline__ v8f v8zero() {
  v8f z;
#pragma unroll
  for (int j = 0; j < 8; ++j) z[j] = 0.f;
  return z;
}

// ---------------- x: fp32 -> bf16 (8 elems / thread) ----------------
__global__ void pack_x_kernel(const float* __restrict__ x, u16* __restrict__ o, int n8) {
  int i = blockIdx.x * blockDim.x + threadIdx.x;
  if (i >= n8) return;
  const float* p = x + ((size_t)i << 3);
  v4u r;
#pragma unroll
  for (int j = 0; j < 4; ++j) {
    unsigned lo = f2bf(p[2 * j]);
    unsigned hi = f2bf(p[2 * j + 1]);
    r[j] = lo | (hi << 16);
  }
  *(v4u*)(o + ((size_t)i << 3)) = r;
}

// ---- pack combined weights [Whh | Wih] into WMMA B-fragment layout ----
// Fragment (nt, kt): 32 lanes x 16 bf16 (32B/lane). B = W^T; lane L holds
// column n = nt*16 + (L&15); element i -> k_local = (i&7) + (i>=8?16:0) + (L>=16?8:0).
__global__ void pack_w_kernel(const float* __restrict__ Whh, const float* __restrict__ Wih,
                              u16* __restrict__ out, int Kh, int Ki, int kTiles, int total) {
  int idx = blockIdx.x * blockDim.x + threadIdx.x;
  if (idx >= total) return;
  int i  = idx & 15;
  int L  = (idx >> 4) & 31;
  int fr = idx >> 9;
  int kt = fr % kTiles;
  int nt = fr / kTiles;
  int n  = nt * 16 + (L & 15);
  int kl = (i & 7) + ((i >> 3) << 4) + ((L >> 4) << 3);
  int k  = kt * 32 + kl;
  float v = (k < Kh) ? Whh[(size_t)n * Kh + k] : Wih[(size_t)n * Ki + (k - Kh)];
  out[idx] = f2bf(v);
}

// ---------------- layer 1 recurrence ----------------
__launch_bounds__(256, 1)
__global__ void lstm1_kernel(const u16* __restrict__ xbf, const u16* __restrict__ wp,
                             const float* __restrict__ bih, const float* __restrict__ bhh,
                             u16* __restrict__ h1out) {
  __shared__ __align__(16) u16 hbuf[2][MTILE * LDST1];
  const int tid   = threadIdx.x;
  const int lane  = tid & 31;
  const int wave  = tid >> 5;
  const int b0    = blockIdx.x * MTILE;
  const int r     = lane & 15;            // A-matrix row (batch row in tile) / C column
  const int koff  = (lane >> 4) << 3;     // K sub-offset per lane half (0 or 8)
  const int rbase = koff;                 // C rows 0..7 (lane<16) or 8..15
  const int hb    = wave * 32;            // 32 h-columns owned by this wave

  {
    u16* hz = &hbuf[0][0];
    for (int idx = tid; idx < 2 * MTILE * LDST1; idx += 256) hz[idx] = 0;
  }
  __syncthreads();

  float bias[4][2];
#pragma unroll
  for (int q = 0; q < 4; ++q)
#pragma unroll
    for (int s = 0; s < 2; ++s) {
      int n = q * HID1 + hb + s * 16 + r;
      bias[q][s] = bih[n] + bhh[n];
    }

  float cst[2][8];
#pragma unroll
  for (int s = 0; s < 2; ++s)
#pragma unroll
    for (int j = 0; j < 8; ++j) cst[s][j] = 0.f;

  const u16* xbase = xbf + (size_t)(b0 + r) * (TSTEPS * NFEAT);

#pragma clang loop unroll(disable)
  for (int t = 0; t < TSTEPS; ++t) {
    const int cur = t & 1, nxt = cur ^ 1;

    // Opaque zero SGPR offset: loads below cannot be proven loop-invariant
    // (no LICM -> no 640-VGPR spill), but pointer provenance is preserved so
    // they lower to global_load_b128 (saddr + imm), not flat_load.
    unsigned toff = 0;
    asm volatile("" : "+s"(toff));
    const u16* wci = wp + toff;

    v8f acc[4][2];
#pragma unroll
    for (int q = 0; q < 4; ++q)
#pragma unroll
      for (int s = 0; s < 2; ++s) acc[q][s] = v8zero();

    // K tiles 0..7: h_{t-1} from LDS (ds_load_b128 A-fragments).
    // Unroll x2: two A-fragments + 16 B-fragment loads per iteration give the
    // scheduler deeper load/WMMA overlap without spilling.
#pragma unroll 2
    for (int kt = 0; kt < 8; ++kt) {
      FragU a;
      const u16* hr = &hbuf[cur][r * LDST1 + kt * 32 + koff];
      a.u[0] = *(const v4u*)hr;
      a.u[1] = *(const v4u*)(hr + 16);
#pragma unroll
      for (int q = 0; q < 4; ++q)
#pragma unroll
        for (int s = 0; s < 2; ++s) {
          const int nt = q * 16 + wave * 2 + s;
          const u16* bp = wci + (((size_t)(nt * KT1 + kt) << 9) + (lane << 4));
          FragU b;
          b.u[0] = ((const v4u*)bp)[0];
          b.u[1] = ((const v4u*)bp)[1];
          acc[q][s] = __builtin_amdgcn_wmma_f32_16x16x32_bf16(
              false, a.bf, false, b.bf, (short)0, acc[q][s], false, false);
        }
    }
    // K tiles 8..9: x_t from global bf16
#pragma clang loop unroll(disable)
    for (int kx = 0; kx < 2; ++kx) {
      FragU a;
      const u16* xr = xbase + t * NFEAT + kx * 32 + koff;
      a.u[0] = *(const v4u*)xr;
      a.u[1] = *(const v4u*)(xr + 16);
#pragma unroll
      for (int q = 0; q < 4; ++q)
#pragma unroll
        for (int s = 0; s < 2; ++s) {
          const int nt = q * 16 + wave * 2 + s;
          const u16* bp = wci + (((size_t)(nt * KT1 + (8 + kx)) << 9) + (lane << 4));
          FragU b;
          b.u[0] = ((const v4u*)bp)[0];
          b.u[1] = ((const v4u*)bp)[1];
          acc[q][s] = __builtin_amdgcn_wmma_f32_16x16x32_bf16(
              false, a.bf, false, b.bf, (short)0, acc[q][s], false, false);
        }
    }

    // cell update (fp32), write h to LDS (next step) + global (layer-2 input)
#pragma unroll
    for (int s = 0; s < 2; ++s) {
      const int col = hb + s * 16 + r;
#pragma unroll
      for (int j = 0; j < 8; ++j) {
        float iv = sigf(acc[0][s][j] + bias[0][s]);
        float fv = sigf(acc[1][s][j] + bias[1][s]);
        float gv = tanhf(acc[2][s][j] + bias[2][s]);
        float ov = sigf(acc[3][s][j] + bias[3][s]);
        float cn = fv * cst[s][j] + iv * gv;
        cst[s][j] = cn;
        float hv = ov * tanhf(cn);
        u16 h16 = f2bf(hv);
        int row = rbase + j;
        hbuf[nxt][row * LDST1 + col] = h16;
        h1out[(size_t)(b0 + row) * (TSTEPS * HID1) + (size_t)t * HID1 + col] = h16;
      }
    }
    __syncthreads();
  }
}

// ---------------- layer 2 recurrence ----------------
__launch_bounds__(256, 1)
__global__ void lstm2_kernel(const u16* __restrict__ h1, const u16* __restrict__ wp,
                             const float* __restrict__ bih, const float* __restrict__ bhh,
                             float* __restrict__ out) {
  __shared__ __align__(16) u16 hbuf[2][MTILE * LDST2];
  const int tid   = threadIdx.x;
  const int lane  = tid & 31;
  const int wave  = tid >> 5;
  const int b0    = blockIdx.x * MTILE;
  const int r     = lane & 15;
  const int koff  = (lane >> 4) << 3;
  const int rbase = koff;
  const int hb    = wave * 16;            // 16 h2-columns per wave

  float* outseq  = out;
  float* outlast = out + (size_t)BATCH * TSTEPS * EMB2;

  {
    u16* hz = &hbuf[0][0];
    for (int idx = tid; idx < 2 * MTILE * LDST2; idx += 256) hz[idx] = 0;
  }
  __syncthreads();

  float bias[4];
#pragma unroll
  for (int q = 0; q < 4; ++q) {
    int n = q * EMB2 + hb + r;
    bias[q] = bih[n] + bhh[n];
  }
  float cst[8];
#pragma unroll
  for (int j = 0; j < 8; ++j) cst[j] = 0.f;

  const u16* h1base = h1 + (size_t)(b0 + r) * (TSTEPS * HID1);

#pragma clang loop unroll(disable)
  for (int t = 0; t < TSTEPS; ++t) {
    const int cur = t & 1, nxt = cur ^ 1;

    unsigned toff = 0;
    asm volatile("" : "+s"(toff));   // defeat LICM, keep global provenance
    const u16* wci = wp + toff;

    v8f acc[4];
#pragma unroll
    for (int q = 0; q < 4; ++q) acc[q] = v8zero();

    // K tiles 0..3: h2_{t-1} from LDS
#pragma clang loop unroll(disable)
    for (int kt = 0; kt < 4; ++kt) {
      FragU a;
      const u16* hr = &hbuf[cur][r * LDST2 + kt * 32 + koff];
      a.u[0] = *(const v4u*)hr;
      a.u[1] = *(const v4u*)(hr + 16);
#pragma unroll
      for (int q = 0; q < 4; ++q) {
        const int nt = q * 8 + wave;
        const u16* bp = wci + (((size_t)(nt * KT2 + kt) << 9) + (lane << 4));
        FragU b;
        b.u[0] = ((const v4u*)bp)[0];
        b.u[1] = ((const v4u*)bp)[1];
        acc[q] = __builtin_amdgcn_wmma_f32_16x16x32_bf16(
            false, a.bf, false, b.bf, (short)0, acc[q], false, false);
      }
    }
    // K tiles 4..11: layer-1 output h1[b, t, :] from global bf16 (unroll x2)
#pragma unroll 2
    for (int kx = 0; kx < 8; ++kx) {
      FragU a;
      const u16* xr = h1base + t * HID1 + kx * 32 + koff;
      a.u[0] = *(const v4u*)xr;
      a.u[1] = *(const v4u*)(xr + 16);
#pragma unroll
      for (int q = 0; q < 4; ++q) {
        const int nt = q * 8 + wave;
        const u16* bp = wci + (((size_t)(nt * KT2 + (4 + kx)) << 9) + (lane << 4));
        FragU b;
        b.u[0] = ((const v4u*)bp)[0];
        b.u[1] = ((const v4u*)bp)[1];
        acc[q] = __builtin_amdgcn_wmma_f32_16x16x32_bf16(
            false, a.bf, false, b.bf, (short)0, acc[q], false, false);
      }
    }

    const int col = hb + r;
#pragma unroll
    for (int j = 0; j < 8; ++j) {
      float iv = sigf(acc[0][j] + bias[0]);
      float fv = sigf(acc[1][j] + bias[1]);
      float gv = tanhf(acc[2][j] + bias[2]);
      float ov = sigf(acc[3][j] + bias[3]);
      float cn = fv * cst[j] + iv * gv;
      cst[j] = cn;
      float hv = ov * tanhf(cn);
      int row = rbase + j;
      hbuf[nxt][row * LDST2 + col] = f2bf(hv);
      outseq[(size_t)(b0 + row) * (TSTEPS * EMB2) + (size_t)t * EMB2 + col] = hv;
      if (t == TSTEPS - 1) outlast[(size_t)(b0 + row) * EMB2 + col] = hv;
    }
    __syncthreads();
  }
}

extern "C" void kernel_launch(void* const* d_in, const int* in_sizes, int n_in,
                              void* d_out, int out_size, void* d_ws, size_t ws_size,
                              hipStream_t stream) {
  (void)in_sizes; (void)n_in; (void)out_size; (void)ws_size;
  const float* x    = (const float*)d_in[0];
  const float* Wih1 = (const float*)d_in[1];
  const float* Whh1 = (const float*)d_in[2];
  const float* bih1 = (const float*)d_in[3];
  const float* bhh1 = (const float*)d_in[4];
  const float* Wih2 = (const float*)d_in[5];
  const float* Whh2 = (const float*)d_in[6];
  const float* bih2 = (const float*)d_in[7];
  const float* bhh2 = (const float*)d_in[8];
  float* out = (float*)d_out;

  // Workspace layout (1 KiB-aligned regions):
  //   [0, 16 MiB)            : x in bf16
  //   [16 MiB, +640 KiB)     : layer-1 packed weights [Whh1|Wih1], 1024x320 bf16
  //   [+640 KiB, +384 KiB)   : layer-2 packed weights [Whh2|Wih2], 512x384 bf16
  //   [17 MiB, +64 MiB)      : layer-1 output h1 [B,T,256] bf16
  char* ws  = (char*)d_ws;
  u16* xbf  = (u16*)(ws);
  u16* w1p  = (u16*)(ws + (size_t)16 * 1024 * 1024);
  u16* w2p  = (u16*)(ws + (size_t)16 * 1024 * 1024 + 640 * 1024);
  u16* h1   = (u16*)(ws + (size_t)17 * 1024 * 1024);

  const int n8 = (BATCH * TSTEPS * NFEAT) / 8;            // 1,048,576
  pack_x_kernel<<<n8 / 256, 256, 0, stream>>>(x, xbf, n8);

  const int tot1 = GATE1 * KTOT1;                          // 327,680
  pack_w_kernel<<<(tot1 + 255) / 256, 256, 0, stream>>>(Whh1, Wih1, w1p, HID1, NFEAT, KT1, tot1);
  const int tot2 = GATE2 * KTOT2;                          // 196,608
  pack_w_kernel<<<(tot2 + 255) / 256, 256, 0, stream>>>(Whh2, Wih2, w2p, EMB2, HID1, KT2, tot2);

  lstm1_kernel<<<BATCH / MTILE, 256, 0, stream>>>(xbf, w1p, bih1, bhh1, h1);
  lstm2_kernel<<<BATCH / MTILE, 256, 0, stream>>>(h1, w2p, bih2, bhh2, out);
}
